// E3_TransformerLayer_multi_76639396430367
// MI455X (gfx1250) — compile-verified, matrix operand order checked by reading, not run on hardware
//
#include <hip/hip_runtime.h>
#include <hip/hip_bf16.h>

typedef __attribute__((ext_vector_type(16))) _Float16 v16h;
typedef __attribute__((ext_vector_type(8)))  float    v8f;

// ---------------- LDS layout for the fused edge kernel ----------------
// wf1L : 64 cols x 32 K (f16, K padded 16->32 with zeros)   =   4096 B
// wf2L : 64 cols x 64 K (f16)                               =   8192 B
// wf3L : 64 cols x 64 K (f16)                               =   8192 B
// wf4L : 1536 cols x 64 K (f16)                             = 196608 B
// per wave (x8): actA 16x64 f16 (2048) + actB (2048)
//                + P 16x64 f32 (4096) + tbuf 16x24 f32 (1536) = 9728 B
#define WAVE_SMEM 9728
#define SMEM_BYTES (4096 + 8192 + 8192 + 196608 + 8 * WAVE_SMEM) // 294912 <= 320K

// Fast SiLU: x * rcp(1+exp(-x)) -> v_exp_f32 + v_rcp_f32 + mul
// (avoids the full IEEE division expansion seen in round-1 asm)
__device__ __forceinline__ float siluf(float x) {
  return x * __builtin_amdgcn_rcpf(1.0f + __expf(-x));
}

// Load a 16-f16 WMMA operand fragment: two contiguous 16B runs at p and p+16
// (matches the CDNA5 16-bit A/B layout: lanes 0-15 hold K {b..b+7, b+16..b+23},
//  lanes 16-31 hold K {b+8..b+15, b+24..b+31}; caller folds hi*8 into p).
__device__ __forceinline__ v16h ld_frag(const _Float16* p) {
  v16h f;
  *reinterpret_cast<uint4*>(&f)       = *reinterpret_cast<const uint4*>(p);
  *(reinterpret_cast<uint4*>(&f) + 1) = *reinterpret_cast<const uint4*>(p + 16);
  return f;
}

__device__ __forceinline__ v8f wmma32(v16h a, v16h b, v8f c) {
  return __builtin_amdgcn_wmma_f32_16x16x32_f16(false, a, false, b, (short)0, c,
                                                false, false);
}

__device__ __forceinline__ void wave_lds_fence() {
  __asm__ volatile("s_wait_dscnt 0" ::: "memory");
}

// one 16x64 @ 64x64 GEMM + SiLU, act buffers are wave-private LDS
__device__ __forceinline__ void gemm64_silu(const _Float16* aIn, const _Float16* wL,
                                            _Float16* aOut, int rc, int hiK, int mb) {
  v16h a0 = ld_frag(aIn + rc * 64 + hiK * 8);
  v16h a1 = ld_frag(aIn + rc * 64 + 32 + hiK * 8);
#pragma unroll
  for (int j = 0; j < 4; ++j) {
    v16h b0 = ld_frag(wL + (16 * j + rc) * 64 + hiK * 8);
    v16h b1 = ld_frag(wL + (16 * j + rc) * 64 + 32 + hiK * 8);
    v8f c = {0, 0, 0, 0, 0, 0, 0, 0};
    c = wmma32(a0, b0, c);
    c = wmma32(a1, b1, c);
#pragma unroll
    for (int rr = 0; rr < 8; ++rr)
      aOut[(rr + mb) * 64 + 16 * j + rc] = (_Float16)siluf(c[rr]);
  }
  wave_lds_fence();
}

// ---------------------------------------------------------------------
__global__ void zero_kernel(float* out, float* counts, int nOut, int N) {
  int i = blockIdx.x * blockDim.x + threadIdx.x;
  if (i < nOut) out[i] = 0.0f;
  if (i < N) counts[i] = 0.0f;
}

__global__ void node_mlp_kernel(const float* __restrict__ emb_table,
                                const float* __restrict__ W1, const float* __restrict__ b1,
                                const float* __restrict__ W2, const float* __restrict__ b2,
                                const float* __restrict__ W3, const float* __restrict__ b3,
                                const int* __restrict__ A, float* __restrict__ Ai, int N) {
  __shared__ float sE[160], sW1[1024], sB1[64], sW2[2048], sB2[32], sW3[256], sB3[8];
  int tid = threadIdx.x;
  for (int i = tid; i < 160; i += 256) sE[i] = emb_table[i];
  for (int i = tid; i < 1024; i += 256) sW1[i] = W1[i];
  for (int i = tid; i < 64; i += 256) sB1[i] = b1[i];
  for (int i = tid; i < 2048; i += 256) sW2[i] = W2[i];
  for (int i = tid; i < 32; i += 256) sB2[i] = b2[i];
  for (int i = tid; i < 256; i += 256) sW3[i] = W3[i];
  for (int i = tid; i < 8; i += 256) sB3[i] = b3[i];
  __syncthreads();
  int n = blockIdx.x * 256 + tid;
  if (n >= N) return;
  int a = A[n];
  float e[16];
#pragma unroll
  for (int i = 0; i < 16; ++i) e[i] = sE[a * 16 + i];
  float h1[64];
#pragma unroll 8
  for (int j = 0; j < 64; ++j) {
    float s = sB1[j];
#pragma unroll
    for (int i = 0; i < 16; ++i) s += e[i] * sW1[i * 64 + j];
    h1[j] = siluf(s);
  }
  float h2[32];
#pragma unroll 4
  for (int j = 0; j < 32; ++j) {
    float s = sB2[j];
#pragma unroll
    for (int i = 0; i < 64; ++i) s += h1[i] * sW2[i * 32 + j];
    h2[j] = siluf(s);
  }
#pragma unroll
  for (int o = 0; o < 8; ++o) {
    float s = sB3[o];
#pragma unroll
    for (int i = 0; i < 32; ++i) s += h2[i] * sW3[i * 8 + o];
    Ai[(size_t)n * 8 + o] = s;
  }
}

__global__ __launch_bounds__(256, 1) void edge_kernel(
    const float* __restrict__ pos, const float* __restrict__ edge_shifts,
    const float* __restrict__ cell, const float* __restrict__ Wf1,
    const float* __restrict__ Wf2, const float* __restrict__ Wf3,
    const float* __restrict__ Wf4, const float* __restrict__ Ai,
    float* __restrict__ counts, const int* __restrict__ edge_src,
    const int* __restrict__ edge_dst, const int* __restrict__ batch,
    float* __restrict__ out, int E) {
  extern __shared__ char smem[];
  _Float16* wf1L = (_Float16*)smem;       // 64*32
  _Float16* wf2L = wf1L + 64 * 32;        // 64*64
  _Float16* wf3L = wf2L + 64 * 64;        // 64*64
  _Float16* wf4L = wf3L + 64 * 64;        // 1536*64
  char* waveMem = (char*)(wf4L + 1536 * 64) + (threadIdx.x >> 5) * WAVE_SMEM;
  _Float16* actA = (_Float16*)waveMem;    // 16*64 f16
  _Float16* actB = actA + 1024;           // 16*64 f16
  float* P = (float*)(actB + 1024);       // 16*64 f32 (outer product Ai_src x Ai_dst)
  float* tbuf = P + 1024;                 // 16*24 f32

  int tid = threadIdx.x;
  // Stage filter weights into LDS, f16, column-major [N][K] for B fragments.
  for (int i = tid; i < 64 * 32; i += 256) {
    int n = i >> 5, k = i & 31;
    wf1L[i] = (k < 16) ? (_Float16)Wf1[k * 64 + n] : (_Float16)0.0f;
  }
  for (int i = tid; i < 64 * 64; i += 256) {
    int n = i >> 6, k = i & 63;
    wf2L[i] = (_Float16)Wf2[k * 64 + n];
    wf3L[i] = (_Float16)Wf3[k * 64 + n];
  }
  for (int i = tid; i < 1536 * 64; i += 256) {
    int n = i >> 6, k = i & 63;
    wf4L[i] = (_Float16)Wf4[k * 1536 + n];
  }
  __syncthreads();

  int lane = tid & 31;
  int wave = tid >> 5;
  int rc = lane & 15;        // row (edge-in-tile) for A, col for B/C
  int hiK = lane >> 4;       // which K half-set this lane owns
  int mb = hiK * 8;          // C fragment row offset
  int T = (E + 15) >> 4;

  for (int t0 = blockIdx.x * 8 + wave; t0 < T; t0 += gridDim.x * 8) {
    // ---------------- Phase A: per-edge scalars (lanes 0-15) ----------
    float shv[9] = {0, 0, 0, 0, 0, 0, 0, 0, 0};
    int dstN = 0;
    bool valid = false;
    if (lane < 16) {
      int e = t0 * 16 + rc;
      valid = e < E;
      int ee = valid ? e : 0;
      int src = edge_src[ee], dst = edge_dst[ee];
      dstN = dst;
      int b = batch[src];
      float s0 = edge_shifts[(size_t)ee * 3 + 0];
      float s1 = edge_shifts[(size_t)ee * 3 + 1];
      float s2 = edge_shifts[(size_t)ee * 3 + 2];
      const float* cb = cell + (size_t)b * 9;
      float vx = pos[(size_t)dst * 3 + 0] - pos[(size_t)src * 3 + 0] +
                 s0 * cb[0] + s1 * cb[3] + s2 * cb[6];
      float vy = pos[(size_t)dst * 3 + 1] - pos[(size_t)src * 3 + 1] +
                 s0 * cb[1] + s1 * cb[4] + s2 * cb[7];
      float vz = pos[(size_t)dst * 3 + 2] - pos[(size_t)src * 3 + 2] +
                 s0 * cb[2] + s1 * cb[5] + s2 * cb[8];
      float r = sqrtf(vx * vx + vy * vy + vz * vz);
      float inv = 1.0f / (r + 1e-9f);
      float ux = vx * inv, uy = vy * inv, uz = vz * inv;
      const float c3 = 1.7320508075688772f;   // sqrt(3)
      const float c5 = 2.2360679774997896f;   // sqrt(5)
      const float c15 = 3.872983346207417f;   // sqrt(15)
      shv[0] = 1.0f;
      shv[1] = c3 * uy;  shv[2] = c3 * uz;  shv[3] = c3 * ux;
      shv[4] = c15 * ux * uy;  shv[5] = c15 * uy * uz;
      shv[6] = 0.5f * c5 * (3.0f * uz * uz - 1.0f);
      shv[7] = c15 * uz * ux;
      shv[8] = 0.5f * c15 * (ux * ux - uy * uy);
      // radial basis -> A-matrix row (K padded to 32)
      const float step = 5.0f / 17.0f, istep = 17.0f / 5.0f, amp = 4.0f / 1.12f;
#pragma unroll
      for (int k = 0; k < 16; ++k) {
        float d = (r - (float)(k + 1) * step) * istep;
        float v = valid ? __expf(-d * d) * amp : 0.0f;
        actA[rc * 64 + k] = (_Float16)v;
      }
#pragma unroll
      for (int k = 16; k < 32; ++k) actA[rc * 64 + k] = (_Float16)0.0f;
      float as[8], ad[8];
#pragma unroll
      for (int u = 0; u < 8; ++u) {
        as[u] = Ai[(size_t)src * 8 + u];
        ad[u] = Ai[(size_t)dst * 8 + u];
      }
#pragma unroll
      for (int u = 0; u < 8; ++u)
#pragma unroll
        for (int v = 0; v < 8; ++v)
          P[rc * 64 + u * 8 + v] = valid ? as[u] * ad[v] : 0.0f;
      if (valid) atomicAdd(&counts[dst], 1.0f);
    }
    wave_lds_fence();

    // ---------------- G1: emb(16 edges x 16, pad 32) @ Wf1 -> 64 ------
    {
      v16h a = ld_frag(actA + rc * 64 + hiK * 8);
#pragma unroll
      for (int j = 0; j < 4; ++j) {
        v16h b = ld_frag(wf1L + (16 * j + rc) * 32 + hiK * 8);
        v8f c = {0, 0, 0, 0, 0, 0, 0, 0};
        c = wmma32(a, b, c);
#pragma unroll
        for (int rr = 0; rr < 8; ++rr)
          actB[(rr + mb) * 64 + 16 * j + rc] = (_Float16)siluf(c[rr]);
      }
    }
    wave_lds_fence();
    // ---------------- G2, G3: 64 -> 64 -> 64 --------------------------
    gemm64_silu(actB, wf2L, actA, rc, hiK, mb);
    gemm64_silu(actA, wf3L, actB, rc, hiK, mb);

    // ---------------- G4: g(16x64) @ Wf4(64x1536), fused contraction --
    {
      v16h a0 = ld_frag(actB + rc * 64 + hiK * 8);
      v16h a1 = ld_frag(actB + rc * 64 + 32 + hiK * 8);
      float tacc[24];
#pragma unroll
      for (int i = 0; i < 24; ++i) tacc[i] = 0.0f;
      int hi3 = (lane >> 3) & 1;  // lane's uv parity within the fragment
      for (int j2 = 0; j2 < 32; ++j2) {
        int uv = 2 * j2 + hi3;    // (u*8+v) index this lane's column hits
        float pv[8];
#pragma unroll
        for (int rr = 0; rr < 8; ++rr) pv[rr] = P[(rr + mb) * 64 + uv];
#pragma unroll
        for (int l = 0; l < 3; ++l) {
          int col = 16 * (l * 32 + j2) + rc;
          v16h b0 = ld_frag(wf4L + col * 64 + hiK * 8);
          v16h b1 = ld_frag(wf4L + col * 64 + 32 + hiK * 8);
          v8f c = {0, 0, 0, 0, 0, 0, 0, 0};
          c = wmma32(a0, b0, c);
          c = wmma32(a1, b1, c);
#pragma unroll
          for (int rr = 0; rr < 8; ++rr) tacc[l * 8 + rr] += pv[rr] * c[rr];
        }
      }
      // combine lane pairs (n, n^8): same o = lane&7, complementary uv half
#pragma unroll
      for (int l = 0; l < 3; ++l)
#pragma unroll
        for (int rr = 0; rr < 8; ++rr) {
          float s = tacc[l * 8 + rr] + __shfl_xor(tacc[l * 8 + rr], 8, 32);
          if ((lane & 8) == 0)
            tbuf[(rr + mb) * 24 + l * 8 + (lane & 7)] = s;
        }
    }
    wave_lds_fence();

    // ---------------- scatter: out[dst] += sh ⊗ t * nrm ---------------
    if (valid) {
      const float nrm = 0.125f;  // 1/sqrt(OUT*OUT)
      float* o = out + (size_t)dstN * 72;
      const float* tm = tbuf + rc * 24;
#pragma unroll
      for (int oo = 0; oo < 8; ++oo) atomicAdd(o + oo, tm[oo] * nrm);
#pragma unroll
      for (int oo = 0; oo < 8; ++oo)
#pragma unroll
        for (int m2 = 0; m2 < 3; ++m2)
          atomicAdd(o + 8 + oo * 3 + m2, tm[8 + oo] * shv[1 + m2] * nrm);
#pragma unroll
      for (int oo = 0; oo < 8; ++oo)
#pragma unroll
        for (int m2 = 0; m2 < 5; ++m2)
          atomicAdd(o + 32 + oo * 5 + m2, tm[16 + oo] * shv[4 + m2] * nrm);
    }
    wave_lds_fence();
  }
}

__global__ void norm_kernel(float* out, const float* counts, int N) {
  int i = blockIdx.x * blockDim.x + threadIdx.x;
  if (i < N * 72) out[i] /= fmaxf(counts[i / 72], 1.0f);
}

// ---------------------------------------------------------------------
extern "C" void kernel_launch(void* const* d_in, const int* in_sizes, int n_in,
                              void* d_out, int out_size, void* d_ws, size_t ws_size,
                              hipStream_t stream) {
  const float* pos = (const float*)d_in[0];
  const float* edge_shifts = (const float*)d_in[1];
  const float* cell = (const float*)d_in[2];
  const float* emb_table = (const float*)d_in[3];
  const float* W1 = (const float*)d_in[4];
  const float* b1 = (const float*)d_in[5];
  const float* W2 = (const float*)d_in[6];
  const float* b2 = (const float*)d_in[7];
  const float* W3 = (const float*)d_in[8];
  const float* b3 = (const float*)d_in[9];
  const float* Wf1 = (const float*)d_in[10];
  const float* Wf2 = (const float*)d_in[11];
  const float* Wf3 = (const float*)d_in[12];
  const float* Wf4 = (const float*)d_in[13];
  const int* A = (const int*)d_in[14];
  const int* batch = (const int*)d_in[15];
  const int* edge_src = (const int*)d_in[16];
  const int* edge_dst = (const int*)d_in[17];

  int N = in_sizes[14];
  int E = in_sizes[16];
  float* out = (float*)d_out;
  float* Ai = (float*)d_ws;                 // N*8 floats
  float* counts = Ai + (size_t)N * 8;       // N floats

  int nOut = N * 72;
  zero_kernel<<<(nOut + 255) / 256, 256, 0, stream>>>(out, counts, nOut, N);
  node_mlp_kernel<<<(N + 255) / 256, 256, 0, stream>>>(emb_table, W1, b1, W2, b2,
                                                       W3, b3, A, Ai, N);
  int T = (E + 15) / 16;
  int blocks = (T + 7) / 8;
  if (blocks > 512) blocks = 512;
  if (blocks < 1) blocks = 1;
  (void)hipFuncSetAttribute((const void*)edge_kernel,
                            hipFuncAttributeMaxDynamicSharedMemorySize, SMEM_BYTES);
  edge_kernel<<<blocks, 256, SMEM_BYTES, stream>>>(pos, edge_shifts, cell, Wf1, Wf2,
                                                   Wf3, Wf4, Ai, counts, edge_src,
                                                   edge_dst, batch, out, E);
  norm_kernel<<<(nOut + 255) / 256, 256, 0, stream>>>(out, counts, N);
}
